// GCN_dgl_2817498546215
// MI455X (gfx1250) — matrix operand order repaired and checked by reading, not compiled, not run on hardware
//
#include <hip/hip_runtime.h>
#include <hip/hip_bf16.h>

typedef __attribute__((ext_vector_type(16))) __bf16 v16bf;
typedef __attribute__((ext_vector_type(8)))  __bf16 v8bf;
typedef __attribute__((ext_vector_type(8)))  float  v8f;

static __device__ __forceinline__ void atomic_add_f32(float* p, float v) {
    __hip_atomic_fetch_add(p, v, __ATOMIC_RELAXED, __HIP_MEMORY_SCOPE_AGENT);
}

// ---------------------------------------------------------------- utilities
__global__ void gcn_zero_f32(float* __restrict__ p, long n) {
    long i = (long)blockIdx.x * blockDim.x + threadIdx.x;
    long stride = (long)gridDim.x * blockDim.x;
    for (; i < n; i += stride) p[i] = 0.0f;
}

__global__ void gcn_degree(const int* __restrict__ src, const int* __restrict__ dst,
                           float* __restrict__ degO, float* __restrict__ degI, int E) {
    int e = blockIdx.x * blockDim.x + threadIdx.x;
    if (e < E) {
        atomic_add_f32(&degO[src[e]], 1.0f);
        atomic_add_f32(&degI[dst[e]], 1.0f);
    }
}

__global__ void gcn_invsqrt(float* __restrict__ degO, float* __restrict__ degI, int n) {
    int i = blockIdx.x * blockDim.x + threadIdx.x;
    if (i < n) {
        degO[i] = rsqrtf(fmaxf(degO[i], 1.0f));
        degI[i] = rsqrtf(fmaxf(degI[i], 1.0f));
    }
}

// out[i] = bf16(h[i] * invO[row]); input feature dim is always 128 (shift 7)
__global__ void gcn_scale_cvt(const float* __restrict__ h, const float* __restrict__ invO,
                              __bf16* __restrict__ out, long n) {
    long i = (long)blockIdx.x * blockDim.x + threadIdx.x;
    if (i < n) {
        long row = i >> 7;
        out[i] = (__bf16)(h[i] * invO[row]);
    }
}

// Pack W (f32, [128 x Dout] row-major) into per-lane-contiguous WMMA B fragments.
// Fragment index = (nt*4 + kt)*32 + lane; 16 bf16 per lane.
// B layout (ISA 7.12.2): lane l -> col = l&15, half = l>>4; element e -> K = kt*32 + 8*half + (e&7) + 16*(e>>3).
__global__ void gcn_pack_b(const float* __restrict__ W, __bf16* __restrict__ Bp, int Dout) {
    int t = blockIdx.x * blockDim.x + threadIdx.x;
    int total = (Dout >> 4) * 4 * 32;
    if (t >= total) return;
    int lane = t & 31;
    int kt   = (t >> 5) & 3;
    int nt   = t >> 7;
    int half = lane >> 4;
    int col  = nt * 16 + (lane & 15);
    int kbase = kt * 32 + half * 8;
    __bf16* o = Bp + (size_t)t * 16;
#pragma unroll
    for (int e = 0; e < 16; ++e)
        o[e] = (__bf16)W[(size_t)(kbase + (e & 7) + ((e >> 3) << 4)) * Dout + col];
}

// C[N x DOUT] = A[N x 128](bf16) @ Bpacked.  One wave per 16-row tile of A.
template <int DOUT>
__global__ void gcn_gemm_wmma(const __bf16* __restrict__ A, const __bf16* __restrict__ Bp,
                              float* __restrict__ C, int Mtiles, int N) {
    int wave = blockIdx.x * (blockDim.x >> 5) + (threadIdx.x >> 5);
    if (wave >= Mtiles) return;                 // wave-uniform: EXEC stays all-1s
    int lane = threadIdx.x & 31;
    int half = lane >> 4;
    int arow = wave * 16 + (lane & 15);
    if (arow > N - 1) arow = N - 1;             // clamp (keeps EXEC uniform)
    const __bf16* ap = A + (size_t)arow * 128;

    // A fragments (ISA 7.12.2 16-bit A 16x32): lane holds K = base0..base0+7 and 16+base0..+7
    v16bf afrag[4];
#pragma unroll
    for (int kt = 0; kt < 4; ++kt) {
        v8bf lo = *(const v8bf*)(ap + kt * 32 + half * 8);
        v8bf hi = *(const v8bf*)(ap + kt * 32 + 16 + half * 8);
#pragma unroll
        for (int e = 0; e < 8; ++e) { afrag[kt][e] = lo[e]; afrag[kt][e + 8] = hi[e]; }
    }

    const bool fullTile = (wave * 16 + 16) <= N;   // wave-uniform
    constexpr int NT = DOUT >> 4;
#pragma unroll
    for (int nt = 0; nt < NT; ++nt) {
        // fetch all 4 B fragments first (one load clause), then 4 back-to-back WMMAs
        v16bf bfrag[4];
#pragma unroll
        for (int kt = 0; kt < 4; ++kt)
            bfrag[kt] = *(const v16bf*)(Bp + ((size_t)(nt * 4 + kt) * 32 + lane) * 16);
        v8f acc = {};
#pragma unroll
        for (int kt = 0; kt < 4; ++kt)
            acc = __builtin_amdgcn_wmma_f32_16x16x32_bf16(
                /*neg_a=*/false, afrag[kt], /*neg_b=*/false, bfrag[kt],
                /*c_mod=*/(short)0, acc, /*reuse_a=*/false, /*reuse_b=*/false);

        // C layout: VGPR r -> tile row (r + 8*half), col = lane&15
        float* cb = C + (size_t)wave * 16 * DOUT + nt * 16 + (lane & 15)
                      + (size_t)(8 * half) * DOUT;
        if (fullTile) {
#pragma unroll
            for (int r = 0; r < 8; ++r) cb[(size_t)r * DOUT] = acc[r];
        } else {
#pragma unroll
            for (int r = 0; r < 8; ++r)
                if (wave * 16 + r + 8 * half < N) cb[(size_t)r * DOUT] = acc[r];
        }
    }
}

// one wave per edge: agg[dst] += lin[src] * edge_w  (coalesced across lanes)
template <int D>
__global__ void gcn_scatter(const float* __restrict__ lin, const int* __restrict__ src,
                            const int* __restrict__ dst, const float* __restrict__ ew,
                            float* __restrict__ agg, int E) {
    int wave = blockIdx.x * (blockDim.x >> 5) + (threadIdx.x >> 5);
    if (wave >= E) return;
    int lane = threadIdx.x & 31;
    int s = src[wave], d = dst[wave];
    float w = ew[wave];
    const float* ls = lin + (size_t)s * D;
    float* ad = agg + (size_t)d * D;
#pragma unroll
    for (int it = 0; it < D / 32; ++it) {
        int j = lane + 32 * it;
        atomic_add_f32(&ad[j], ls[j] * w);
    }
}

template <bool RELU>
__global__ void gcn_finalize(const float* __restrict__ agg, const float* __restrict__ invI,
                             const float* __restrict__ bias, float* __restrict__ out,
                             long n, int dshift, int dmask) {
    long i = (long)blockIdx.x * blockDim.x + threadIdx.x;
    if (i < n) {
        long row = i >> dshift;
        int  col = (int)(i & dmask);
        float v = agg[i] * invI[row] + bias[col];
        out[i] = RELU ? fmaxf(v, 0.0f) : v;
    }
}

// ---------------------------------------------------------------- launcher
extern "C" void kernel_launch(void* const* d_in, const int* in_sizes, int n_in,
                              void* d_out, int out_size, void* d_ws, size_t ws_size,
                              hipStream_t stream) {
    (void)n_in; (void)ws_size; (void)out_size;
    const float* x   = (const float*)d_in[0];
    const float* ew  = (const float*)d_in[1];
    const int*   src = (const int*)d_in[2];
    const int*   dst = (const int*)d_in[3];
    const float* W1  = (const float*)d_in[4];
    const float* b1  = (const float*)d_in[5];
    const float* W2  = (const float*)d_in[6];
    const float* b2  = (const float*)d_in[7];
    const float* W3  = (const float*)d_in[8];
    const float* b3  = (const float*)d_in[9];

    const int N    = in_sizes[0] / 128;     // 100000
    const int E    = in_sizes[2];           // 1600000
    const int DOUT = in_sizes[8] / 128;     // 64
    const int dsh3 = 31 - __builtin_clz(DOUT);

    char* ws = (char*)d_ws;
    size_t off = 0;
    auto carve = [&](size_t bytes) -> char* {
        char* p = ws + off;
        off += (bytes + 255) & ~(size_t)255;
        return p;
    };
    float*  invO = (float*)carve((size_t)N * 4);
    float*  invI = (float*)carve((size_t)N * 4);
    float*  bufA = (float*)carve((size_t)N * 128 * 4);
    float*  bufB = (float*)carve((size_t)N * 128 * 4);
    __bf16* hb   = (__bf16*)carve((size_t)N * 128 * 2);
    __bf16* Bp   = (__bf16*)carve((size_t)8 * 4 * 32 * 16 * 2);

    const int TB = 256;
    auto nb = [](long n, int tb) { return (int)((n + tb - 1) / tb); };
    const long NE128 = (long)N * 128;
    const int  Mtiles = (N + 15) / 16;
    const int  gemmBlocks = (Mtiles + 7) / 8;       // 8 waves / block
    const int  scatBlocks = (E + 7) / 8;            // 8 waves / block

    // degrees -> inv sqrt (clamped >= 1)
    gcn_zero_f32<<<nb(N, TB), TB, 0, stream>>>(invO, N);
    gcn_zero_f32<<<nb(N, TB), TB, 0, stream>>>(invI, N);
    gcn_degree<<<nb(E, TB), TB, 0, stream>>>(src, dst, invO, invI, E);
    gcn_invsqrt<<<nb(N, TB), TB, 0, stream>>>(invO, invI, N);

    // ---- layer 1: x(128) -> bufA(128), ReLU
    gcn_scale_cvt<<<nb(NE128, TB), TB, 0, stream>>>(x, invO, hb, NE128);
    gcn_pack_b<<<nb(8 * 4 * 32, TB), TB, 0, stream>>>(W1, Bp, 128);
    gcn_gemm_wmma<128><<<gemmBlocks, TB, 0, stream>>>(hb, Bp, bufA, Mtiles, N);
    gcn_zero_f32<<<nb(NE128, TB), TB, 0, stream>>>(bufB, NE128);
    gcn_scatter<128><<<scatBlocks, TB, 0, stream>>>(bufA, src, dst, ew, bufB, E);
    gcn_finalize<true><<<nb(NE128, TB), TB, 0, stream>>>(bufB, invI, b1, bufA, NE128, 7, 127);

    // ---- layer 2: bufA(128) -> bufA(128), ReLU
    gcn_scale_cvt<<<nb(NE128, TB), TB, 0, stream>>>(bufA, invO, hb, NE128);
    gcn_pack_b<<<nb(8 * 4 * 32, TB), TB, 0, stream>>>(W2, Bp, 128);
    gcn_gemm_wmma<128><<<gemmBlocks, TB, 0, stream>>>(hb, Bp, bufA, Mtiles, N);
    gcn_zero_f32<<<nb(NE128, TB), TB, 0, stream>>>(bufB, NE128);
    gcn_scatter<128><<<scatBlocks, TB, 0, stream>>>(bufA, src, dst, ew, bufB, E);
    gcn_finalize<true><<<nb(NE128, TB), TB, 0, stream>>>(bufB, invI, b2, bufA, NE128, 7, 127);

    // ---- layer 3: bufA(128) -> d_out(DOUT), no activation
    float* out = (float*)d_out;
    const long NEOUT = (long)N * DOUT;
    gcn_scale_cvt<<<nb(NE128, TB), TB, 0, stream>>>(bufA, invO, hb, NE128);
    gcn_pack_b<<<nb((DOUT >> 4) * 4 * 32, TB), TB, 0, stream>>>(W3, Bp, DOUT);
    if (DOUT == 128) {
        gcn_gemm_wmma<128><<<gemmBlocks, TB, 0, stream>>>(hb, Bp, bufB, Mtiles, N);
        gcn_zero_f32<<<nb(NEOUT, TB), TB, 0, stream>>>(out, NEOUT);
        gcn_scatter<128><<<scatBlocks, TB, 0, stream>>>(bufB, src, dst, ew, out, E);
    } else {
        gcn_gemm_wmma<64><<<gemmBlocks, TB, 0, stream>>>(hb, Bp, bufB, Mtiles, N);
        gcn_zero_f32<<<nb(NEOUT, TB), TB, 0, stream>>>(out, NEOUT);
        gcn_scatter<64><<<scatBlocks, TB, 0, stream>>>(bufB, src, dst, ew, out, E);
    }
    gcn_finalize<false><<<nb(NEOUT, TB), TB, 0, stream>>>(out, invI, b3, out, NEOUT, dsh3, DOUT - 1);
}
